// _CondCouplingLayer_65893388255721
// MI455X (gfx1250) — matrix operand order repaired and checked by reading, not compiled, not run on hardware
//
#include <hip/hip_runtime.h>
#include <hip/hip_bf16.h>

typedef _Float16 h16;
typedef __attribute__((ext_vector_type(16))) _Float16 v16h;
typedef __attribute__((ext_vector_type(8)))  _Float16 v8h;
typedef __attribute__((ext_vector_type(4)))  _Float16 v4h;
typedef __attribute__((ext_vector_type(2)))  _Float16 v2h;
typedef __attribute__((ext_vector_type(8)))  float v8f;
typedef __attribute__((ext_vector_type(4)))  float v4f;
typedef __attribute__((ext_vector_type(2)))  float v2f;

#define BATCH   131072
#define DDIM    256
#define HDIM    320      // RNN + COND
#define HID     128
#define L1K     14       // 448 / 32 K-chunks for layer 1
#define WAVES   4
#define BLOCKS  512
#define ITERS   4        // 8192 tiles / (512 blocks * 4 waves)

// LDS row pitches chosen so 16 row-strided lanes hit disjoint 4-bank windows
#define ZP 456           // z buffer pitch (halves), 448 cols + pad
#define AP 136           // activation pitch (halves), 128 + pad
#define XP 132           // x fixed/free pitch (floats), 128 + pad

// f16 B-fragment weight layout in d_ws (halves)
#define W1F_OFF 0
#define W2F_OFF (112*512)           // 14*8 tiles
#define W3F_OFF ((112+32)*512)      // + 4*8 tiles
#define TOT_FRAG ((112+32+64)*512)  // + 4*16 tiles = 106496 halves = 208KB

// ---------------------------------------------------------------------------
// Prep: convert f32 weights into f16 WMMA B-fragments (ISA 7.12.2 layout):
// tile (kc,nt): lane l holds W[k0 + 16*(e>>3) + 8*(l>>4) + (e&7)][n0 + (l&15)], e = 0..15,
// stored contiguously so the main kernel does one coalesced 32B load per lane.
// ---------------------------------------------------------------------------
__global__ __launch_bounds__(256) void pack_weights(
    const float* __restrict__ W1, const float* __restrict__ W2,
    const float* __restrict__ W3, h16* __restrict__ out) {
  int idx = blockIdx.x * 256 + threadIdx.x;
  if (idx >= TOT_FRAG) return;
  int tile   = idx >> 9;
  int within = idx & 511;
  int lane   = within >> 4;
  int e      = within & 15;
  int khalf  = ((e >> 3) << 4) + ((lane >> 4) << 3) + (e & 7);   // 16*(e/8) + 8*(l/16) + e%8
  int nl     = lane & 15;
  float v;
  if (tile < 112) {                       // W1: [448 x 128], 14 kc x 8 nt
    int kc = tile >> 3, nt = tile & 7;
    v = W1[(size_t)(kc * 32 + khalf) * 128 + nt * 16 + nl];
  } else if (tile < 144) {                // W2: [128 x 128], 4 kc x 8 nt
    int t = tile - 112; int kc = t >> 3, nt = t & 7;
    v = W2[(size_t)(kc * 32 + khalf) * 128 + nt * 16 + nl];
  } else {                                // W3: [128 x 256], 4 kc x 16 nt
    int t = tile - 144; int kc = t >> 4, nt = t & 15;
    v = W3[(size_t)(kc * 32 + khalf) * 256 + nt * 16 + nl];
  }
  out[idx] = (h16)v;
}

// A-fragment (16x32 f16, ISA 7.12.2): per lane, halves K = k0+8*hi .. +7 and
// K = k0+16+8*hi .. +7 are contiguous -> two ds_load_b128.
__device__ inline v16h load_afrag(const h16* p) {
  v8h lo = *(const v8h*)p;
  v8h hh = *(const v8h*)(p + 16);
  v16h a;
#pragma unroll
  for (int i = 0; i < 8; ++i) { a[i] = lo[i]; a[8 + i] = hh[i]; }
  return a;
}

// B-fragment: one coalesced 32B/lane read of the pre-swizzled f16 weights.
__device__ inline v16h load_bfrag(const h16* base, int tile, int l) {
  return *(const v16h*)(base + (size_t)tile * 512 + l * 16);
}

__device__ inline v8f wmma_f16(v16h a, v16h b, v8f c) {
  return __builtin_amdgcn_wmma_f32_16x16x32_f16(
      false, a, false, b, (short)0, c, false, false);
}

// ---------------------------------------------------------------------------
// Main: each wave computes a 16-row tile end-to-end.
// ---------------------------------------------------------------------------
__global__ __launch_bounds__(WAVES * 32) void coupling_main(
    const float* __restrict__ x, const float* __restrict__ hin,
    const float* __restrict__ b1, const float* __restrict__ b2,
    const float* __restrict__ b3, const h16* __restrict__ wf,
    float* __restrict__ y, float* __restrict__ logdet) {
  __shared__ h16   s_z[WAVES][16 * ZP];   // z = [x_even | h] as f16
  __shared__ h16   s_a[WAVES][16 * AP];   // layer-1 activations f16
  __shared__ float s_xf[WAVES][16 * XP];  // x even cols f32 (pass-through)
  __shared__ float s_xr[WAVES][16 * XP];  // x odd cols f32

  const int w  = threadIdx.x >> 5;
  const int l  = threadIdx.x & 31;
  const int ln = l & 15;
  const int hi = l >> 4;

#pragma unroll 1
  for (int it = 0; it < ITERS; ++it) {
    // Launder a zero offset through an SGPR each iteration: keeps the weight
    // pointers in addrspace(1) (-> global_load_b128, LOADcnt only) while
    // preventing LICM from hoisting the B-fragment loads across tiles
    // (which previously caused mass VGPR spills).
    size_t zofs = 0;
    asm volatile("" : "+s"(zofs));
    const h16* w1f = wf + W1F_OFF + zofs;
    const h16* w2f = wf + W2F_OFF + zofs;
    const h16* w3f = wf + W3F_OFF + zofs;

    const int  tile = (blockIdx.x * WAVES + w) + it * (BLOCKS * WAVES);
    const long row0 = (long)tile * 16;

    // ---- stage x: 16 rows x 256 f32, coalesced b128 loads, deinterleave ----
#pragma unroll 4
    for (int ii = 0; ii < 32; ++ii) {
      int i = l + 32 * ii;                 // float4 index, 0..1023
      int r = i >> 6, c4 = i & 63;
      v4f v = *(const v4f*)(x + (row0 + r) * DDIM + 4 * c4);
      v2h ze; ze[0] = (h16)v.x; ze[1] = (h16)v.z;
      *(v2h*)&s_z[w][r * ZP + 2 * c4] = ze;
      v2f xe; xe[0] = v.x; xe[1] = v.z;
      *(v2f*)&s_xf[w][r * XP + 2 * c4] = xe;
      v2f xo; xo[0] = v.y; xo[1] = v.w;
      *(v2f*)&s_xr[w][r * XP + 2 * c4] = xo;
    }
    // ---- stage h: 16 rows x 320 f32 -> f16 into z cols 128..447 ----
#pragma unroll 4
    for (int ii = 0; ii < 40; ++ii) {
      int i = l + 32 * ii;                 // 0..1279
      int r = i / 80, c4 = i % 80;
      v4f v = *(const v4f*)(hin + (row0 + r) * HDIM + 4 * c4);
      v4h hv; hv[0] = (h16)v.x; hv[1] = (h16)v.y; hv[2] = (h16)v.z; hv[3] = (h16)v.w;
      *(v4h*)&s_z[w][r * ZP + 128 + 4 * c4] = hv;
    }
    __syncthreads();

    // ---- layer 1: z[16x448] @ W1 -> relu -> a[16x128] ----
#pragma unroll 1
    for (int nt = 0; nt < 8; ++nt) {
      v8f acc = {};
#pragma unroll
      for (int kc = 0; kc < L1K; ++kc) {
        v16h a = load_afrag(&s_z[w][ln * ZP + kc * 32 + 8 * hi]);
        v16h b = load_bfrag(w1f, kc * 8 + nt, l);
        acc = wmma_f16(a, b, acc);
      }
      float bias = b1[nt * 16 + ln];
#pragma unroll
      for (int i = 0; i < 8; ++i) {
        float f = fmaxf(acc[i] + bias, 0.0f);
        s_a[w][(i + 8 * hi) * AP + nt * 16 + ln] = (h16)f;
      }
    }
    __syncthreads();

    // ---- layer 2: a @ W2 -> relu (result reuses s_z region) ----
    h16* a3buf = &s_z[w][0];
#pragma unroll 1
    for (int nt = 0; nt < 8; ++nt) {
      v8f acc = {};
#pragma unroll
      for (int kc = 0; kc < 4; ++kc) {
        v16h a = load_afrag(&s_a[w][ln * AP + kc * 32 + 8 * hi]);
        v16h b = load_bfrag(w2f, kc * 8 + nt, l);
        acc = wmma_f16(a, b, acc);
      }
      float bias = b2[nt * 16 + ln];
#pragma unroll
      for (int i = 0; i < 8; ++i) {
        float f = fmaxf(acc[i] + bias, 0.0f);
        a3buf[(i + 8 * hi) * AP + nt * 16 + ln] = (h16)f;
      }
    }
    __syncthreads();

    // ---- layer 3 + coupling: s/t tiles share the C layout -> pure VALU ----
    float ldp[8];
#pragma unroll
    for (int i = 0; i < 8; ++i) ldp[i] = 0.0f;
#pragma unroll 1
    for (int nt = 0; nt < 8; ++nt) {
      v8f sacc = {}, tacc = {};
#pragma unroll
      for (int kc = 0; kc < 4; ++kc) {
        v16h a  = load_afrag(&a3buf[ln * AP + kc * 32 + 8 * hi]);
        v16h bs = load_bfrag(w3f, kc * 16 + nt, l);
        v16h bt = load_bfrag(w3f, kc * 16 + 8 + nt, l);
        sacc = wmma_f16(a, bs, sacc);
        tacc = wmma_f16(a, bt, tacc);
      }
      float bsv = b3[nt * 16 + ln];
      float btv = b3[128 + nt * 16 + ln];
#pragma unroll
      for (int i = 0; i < 8; ++i) {
        int m = i + 8 * hi;
        float s = tanhf(sacc[i] + bsv);
        ldp[i] += s;
        float t = tacc[i] + btv;
        float yo = s_xr[w][m * XP + nt * 16 + ln] * __expf(s) + t;
        v2f o; o[0] = s_xf[w][m * XP + nt * 16 + ln]; o[1] = yo;
        *(v2f*)(y + (row0 + m) * DDIM + 2 * (nt * 16 + ln)) = o;  // {even,odd}
      }
    }
    // log_det: butterfly-reduce over the 16-lane N dimension
#pragma unroll
    for (int off = 1; off < 16; off <<= 1) {
#pragma unroll
      for (int i = 0; i < 8; ++i)
        ldp[i] += __shfl_xor(ldp[i], off, 32);
    }
    if (ln == 0) {
#pragma unroll
      for (int i = 0; i < 8; ++i)
        logdet[row0 + i + 8 * hi] = ldp[i];
    }
    __syncthreads();
  }
}

extern "C" void kernel_launch(void* const* d_in, const int* in_sizes, int n_in,
                              void* d_out, int out_size, void* d_ws, size_t ws_size,
                              hipStream_t stream) {
  const float* x  = (const float*)d_in[0];
  const float* h  = (const float*)d_in[1];
  const float* W1 = (const float*)d_in[2];
  const float* b1 = (const float*)d_in[3];
  const float* W2 = (const float*)d_in[4];
  const float* b2 = (const float*)d_in[5];
  const float* W3 = (const float*)d_in[6];
  const float* b3 = (const float*)d_in[7];
  float* y      = (float*)d_out;
  float* logdet = y + (size_t)BATCH * DDIM;
  h16* wf = (h16*)d_ws;

  pack_weights<<<(TOT_FRAG + 255) / 256, 256, 0, stream>>>(W1, W2, W3, wf);
  coupling_main<<<BLOCKS, WAVES * 32, 0, stream>>>(x, h, b1, b2, b3, wf, y, logdet);
}
